// Attention_38740605010293
// MI455X (gfx1250) — matrix operand order repaired and checked by reading, not compiled, not run on hardware
//
#include <hip/hip_runtime.h>

typedef __attribute__((ext_vector_type(16))) __bf16 v16bf;
typedef __attribute__((ext_vector_type(8)))  float  v8f;

#if defined(__gfx1250__) && __has_builtin(__builtin_amdgcn_global_load_async_to_lds_b128)
#define HAVE_ASYNC_LDS 1
#endif

__device__ __forceinline__ __bf16 bits_to_bf16(unsigned short u) {
    union { unsigned short s; __bf16 b; } x; x.s = u; return x.b;
}
__device__ __forceinline__ unsigned short f32_to_bf16_bits(float f) {
    union { float f; unsigned u; } x; x.f = f;
    unsigned r = x.u + 0x7fffu + ((x.u >> 16) & 1u);   // round-to-nearest-even
    return (unsigned short)(r >> 16);
}

#ifdef HAVE_ASYNC_LDS
typedef int v4i_vec __attribute__((vector_size(16)));
typedef __attribute__((address_space(1))) v4i_vec g_v4i;   // global int4
typedef __attribute__((address_space(3))) v4i_vec l_v4i;   // LDS int4

__device__ __forceinline__ void async_copy16(const unsigned short* g, unsigned short* l) {
    __builtin_amdgcn_global_load_async_to_lds_b128((g_v4i*)g, (l_v4i*)l, 0, 0);
}
__device__ __forceinline__ void wait_async0() {
#if __has_builtin(__builtin_amdgcn_s_wait_asynccnt)
    __builtin_amdgcn_s_wait_asynccnt(0);
#else
    asm volatile("s_wait_asynccnt 0x0" ::: "memory");
#endif
}
#endif

// ---- wave32 16-lane-group reductions via ds_swizzle (xor masks stay in half) ----
__device__ __forceinline__ float xmax16(float x) {
    union { float f; int i; } a;
#define SWM(mask) { a.f = x; a.i = __builtin_amdgcn_ds_swizzle(a.i, ((mask) << 10) | 0x1f); x = fmaxf(x, a.f); }
    SWM(1) SWM(2) SWM(4) SWM(8)
#undef SWM
    return x;
}
__device__ __forceinline__ float xsum16(float x) {
    union { float f; int i; } a;
#define SWS(mask) { a.f = x; a.i = __builtin_amdgcn_ds_swizzle(a.i, ((mask) << 10) | 0x1f); x += a.f; }
    SWS(1) SWS(2) SWS(4) SWS(8)
#undef SWS
    return x;
}

// ---------------- f32 -> bf16 conversion ----------------
__global__ void cvt_bf16_kernel(const float* __restrict__ in, unsigned short* __restrict__ out, int n) {
    int i = blockIdx.x * blockDim.x + threadIdx.x;
    if (i < n) out[i] = f32_to_bf16_bits(in[i]);
}

// ---------------- WMMA GEMM: C[M,N] = A[M,K] * B[K,N] ----------------
// Block tile 128x128 (8 waves, wave tile 32x64). K-step 32.
// MODE 0: scatter qkv (N=3072) into q/k/v [b*16][2048][64] bf16, scale q.
// MODE 1: f32 out + bias.
template <int MODE>
__global__ __launch_bounds__(256) void gemm_bf16_kernel(
    const unsigned short* __restrict__ A, const unsigned short* __restrict__ B,
    int N, int K,
    unsigned short* __restrict__ qo, unsigned short* __restrict__ ko, unsigned short* __restrict__ vo,
    float* __restrict__ outF, const float* __restrict__ bias, float qscale)
{
    __shared__ unsigned short As[128 * 40];   // 128 x 32, padded stride 40
    __shared__ unsigned short Bs[32 * 136];   // 32 x 128, padded stride 136

    const int tid  = threadIdx.x;
    const int lane = tid & 31;
    const int wid  = tid >> 5;
    const int ln   = lane & 15;
    const int half = lane >> 4;
    const int waveM = (wid & 3) * 32;
    const int waveN = (wid >> 2) * 64;
    const size_t mBase = (size_t)blockIdx.x * 128;
    const size_t nBase = (size_t)blockIdx.y * 128;

    v8f acc[2][4] = {};

    for (int kb = 0; kb < K; kb += 32) {
#ifdef HAVE_ASYNC_LDS
        // ASYNCcnt-tracked DMA: A tile 128x32 (512 x 16B), B tile 32x128 (256 x 16B)
#pragma unroll
        for (int i = 0; i < 2; ++i) {
            int idx = i * 256 + tid;
            int row = idx >> 2, c = idx & 3;
            async_copy16(A + (mBase + row) * (size_t)K + kb + c * 8, &As[row * 40 + c * 8]);
        }
        {
            int row = tid >> 3, c = tid & 7;
            async_copy16(B + (size_t)(kb + row) * N + nBase + c * 8, &Bs[row * 136 + c * 8]);
        }
        if (kb + 32 < K) {  // prefetch next K-slab into L2
            __builtin_prefetch(A + (mBase + (tid >> 1)) * (size_t)K + kb + 32, 0, 1);
            __builtin_prefetch(B + (size_t)(kb + 32 + (tid >> 3)) * N + nBase, 0, 1);
        }
        wait_async0();
#else
#pragma unroll
        for (int i = 0; i < 8; ++i) {
            int idx = tid * 8 + i;                   // A: 2048 dwords
            int row = idx >> 4, cd = idx & 15;
            unsigned u = *(const unsigned*)(A + (mBase + row) * (size_t)K + kb + cd * 2);
            *(unsigned*)&As[row * 40 + cd * 2] = u;
        }
#pragma unroll
        for (int i = 0; i < 8; ++i) {
            int idx = tid * 8 + i;                   // B: 2048 dwords
            int row = idx >> 6, cd = idx & 63;
            unsigned u = *(const unsigned*)(B + (size_t)(kb + row) * N + nBase + cd * 2);
            *(unsigned*)&Bs[row * 136 + cd * 2] = u;
        }
#endif
        __syncthreads();

        v16bf afrag[2], bfrag[4];
#pragma unroll
        for (int tm = 0; tm < 2; ++tm) {
            int mt = waveM + tm * 16 + ln;
#pragma unroll
            for (int v4 = 0; v4 < 8; ++v4) {
                int kk = ((v4 >= 4) ? 16 : 0) + half * 8 + (v4 & 3) * 2;
                unsigned u = *(const unsigned*)&As[mt * 40 + kk];
                afrag[tm][2 * v4]     = bits_to_bf16((unsigned short)(u & 0xffffu));
                afrag[tm][2 * v4 + 1] = bits_to_bf16((unsigned short)(u >> 16));
            }
        }
#pragma unroll
        for (int tn = 0; tn < 4; ++tn) {
            int nt = waveN + tn * 16 + ln;
#pragma unroll
            for (int e = 0; e < 16; ++e) {
                int kk = ((e >= 8) ? 16 : 0) + half * 8 + (e & 7);
                bfrag[tn][e] = bits_to_bf16(Bs[kk * 136 + nt]);
            }
        }
#pragma unroll
        for (int tm = 0; tm < 2; ++tm)
#pragma unroll
            for (int tn = 0; tn < 4; ++tn)
                acc[tm][tn] = __builtin_amdgcn_wmma_f32_16x16x32_bf16(
                    false, afrag[tm], false, bfrag[tn], (short)0, acc[tm][tn], false, false);
        __syncthreads();
    }

    // epilogue: C layout m = r + 8*half (per VGPR r), n = lane&15
#pragma unroll
    for (int tm = 0; tm < 2; ++tm)
#pragma unroll
        for (int tn = 0; tn < 4; ++tn)
#pragma unroll
            for (int r = 0; r < 8; ++r) {
                size_t gr = mBase + waveM + tm * 16 + r + half * 8;
                size_t gc = nBase + waveN + tn * 16 + ln;
                float val = acc[tm][tn][r];
                if (MODE == 0) {
                    int part = (int)(gc >> 10);
                    int w    = (int)(gc & 1023);
                    int hh   = w >> 6, dd = w & 63;
                    int bb   = (int)(gr >> 11);
                    int ns   = (int)(gr & 2047);
                    size_t dst = (((size_t)(bb * 16 + hh) * 2048) + ns) * 64 + dd;
                    if (part == 0)      qo[dst] = f32_to_bf16_bits(val * qscale);
                    else if (part == 1) ko[dst] = f32_to_bf16_bits(val);
                    else                vo[dst] = f32_to_bf16_bits(val);
                } else {
                    outF[gr * (size_t)N + gc] = val + bias[gc];
                }
            }
}

// ---------------- flash attention: per (b,h), 64 query rows / block ----------------
__global__ __launch_bounds__(128) void attn_kernel(
    const unsigned short* __restrict__ Q, const unsigned short* __restrict__ Km,
    const unsigned short* __restrict__ Vm, unsigned short* __restrict__ Ao)
{
    __shared__ unsigned short Kt[32 * 72];    // K tile 32 x 64, padded
    __shared__ unsigned short Vt[32 * 72];    // V tile 32 x 64, padded
    __shared__ unsigned short P[4][16][32];   // per-wave P tile (bf16)

    const int tid  = threadIdx.x;
    const int lane = tid & 31;
    const int wid  = tid >> 5;
    const int ln   = lane & 15;
    const int half = lane >> 4;
    const int bh   = blockIdx.y;              // b*16 + h
    const int b    = bh >> 4, h = bh & 15;
    const int qRow = blockIdx.x * 64 + wid * 16;
    const size_t base = (size_t)bh * 2048 * 64;

    // Q fragments: A layout, two 32-deep K steps over hd=64
    v16bf aq[2];
#pragma unroll
    for (int ks = 0; ks < 2; ++ks)
#pragma unroll
        for (int v4 = 0; v4 < 8; ++v4) {
            int kk = ks * 32 + ((v4 >= 4) ? 16 : 0) + half * 8 + (v4 & 3) * 2;
            unsigned u = *(const unsigned*)(Q + base + (size_t)(qRow + ln) * 64 + kk);
            aq[ks][2 * v4]     = bits_to_bf16((unsigned short)(u & 0xffffu));
            aq[ks][2 * v4 + 1] = bits_to_bf16((unsigned short)(u >> 16));
        }

    v8f o[4] = {};
    float mi[8], li[8];
#pragma unroll
    for (int r = 0; r < 8; ++r) { mi[r] = -3.0e38f; li[r] = 0.0f; }

    for (int jb = 0; jb < 2048; jb += 32) {
        // stage K/V tiles (shared by all 4 waves)
#ifdef HAVE_ASYNC_LDS
#pragma unroll
        for (int i = 0; i < 2; ++i) {
            int idx = i * 128 + tid;              // 256 x 16B per tile
            int row = idx >> 3, c = idx & 7;
            async_copy16(Km + base + (size_t)(jb + row) * 64 + c * 8, &Kt[row * 72 + c * 8]);
            async_copy16(Vm + base + (size_t)(jb + row) * 64 + c * 8, &Vt[row * 72 + c * 8]);
        }
        wait_async0();
#else
#pragma unroll
        for (int i = 0; i < 4; ++i) {
            int idx = tid * 4 + i;                // 512 dwords per tile
            int row = idx >> 4, cd = idx & 15;
            *(unsigned*)&Kt[row * 72 + cd * 2] =
                *(const unsigned*)(Km + base + (size_t)(jb + row) * 64 + cd * 2);
            *(unsigned*)&Vt[row * 72 + cd * 2] =
                *(const unsigned*)(Vm + base + (size_t)(jb + row) * 64 + cd * 2);
        }
#endif
        __syncthreads();

        // K^T as B fragments from LDS (B layout: K = 16*(e/8) + 8*half + e%8 per 32-step)
        v16bf bk[2][2];
#pragma unroll
        for (int jt = 0; jt < 2; ++jt)
#pragma unroll
            for (int ks = 0; ks < 2; ++ks)
#pragma unroll
                for (int v4 = 0; v4 < 8; ++v4) {
                    int kk = ks * 32 + ((v4 >= 4) ? 16 : 0) + half * 8 + (v4 & 3) * 2;
                    unsigned u = *(const unsigned*)&Kt[(jt * 16 + ln) * 72 + kk];
                    bk[jt][ks][2 * v4]     = bits_to_bf16((unsigned short)(u & 0xffffu));
                    bk[jt][ks][2 * v4 + 1] = bits_to_bf16((unsigned short)(u >> 16));
                }

        v8f s0 = {}, s1 = {};
        s0 = __builtin_amdgcn_wmma_f32_16x16x32_bf16(false, aq[0], false, bk[0][0], (short)0, s0, false, false);
        s0 = __builtin_amdgcn_wmma_f32_16x16x32_bf16(false, aq[1], false, bk[0][1], (short)0, s0, false, false);
        s1 = __builtin_amdgcn_wmma_f32_16x16x32_bf16(false, aq[0], false, bk[1][0], (short)0, s1, false, false);
        s1 = __builtin_amdgcn_wmma_f32_16x16x32_bf16(false, aq[1], false, bk[1][1], (short)0, s1, false, false);

        // online softmax per row (row m = r + 8*half, reduce across 16 lanes)
        float sc[8];
#pragma unroll
        for (int r = 0; r < 8; ++r) {
            float a = s0[r], c = s1[r];
            float rm   = xmax16(fmaxf(a, c));
            float mnew = fmaxf(mi[r], rm);
            float p0 = __expf(a - mnew), p1 = __expf(c - mnew);
            float rs  = xsum16(p0 + p1);
            float scl = __expf(mi[r] - mnew);
            li[r] = li[r] * scl + rs;
            mi[r] = mnew;
            sc[r] = scl;
            s0[r] = p0; s1[r] = p1;
        }
#pragma unroll
        for (int t = 0; t < 4; ++t)
#pragma unroll
            for (int r = 0; r < 8; ++r) o[t][r] *= sc[r];

        // C-layout P -> LDS -> A-layout fragment
#pragma unroll
        for (int r = 0; r < 8; ++r) {
            P[wid][r + half * 8][ln]      = f32_to_bf16_bits(s0[r]);
            P[wid][r + half * 8][16 + ln] = f32_to_bf16_bits(s1[r]);
        }
        asm volatile("s_wait_dscnt 0" ::: "memory");

        v16bf aP;
#pragma unroll
        for (int v4 = 0; v4 < 8; ++v4) {
            int kk = ((v4 >= 4) ? 16 : 0) + half * 8 + (v4 & 3) * 2;
            unsigned u = *(const unsigned*)&P[wid][ln][kk];
            aP[2 * v4]     = bits_to_bf16((unsigned short)(u & 0xffffu));
            aP[2 * v4 + 1] = bits_to_bf16((unsigned short)(u >> 16));
        }

        // O += P @ V   (V tile from LDS as B fragments, 4 hd slices of 16)
#pragma unroll
        for (int t = 0; t < 4; ++t) {
            v16bf bv;
#pragma unroll
            for (int e = 0; e < 16; ++e) {
                int kk = ((e >= 8) ? 16 : 0) + half * 8 + (e & 7);
                bv[e] = bits_to_bf16(Vt[kk * 72 + t * 16 + ln]);
            }
            o[t] = __builtin_amdgcn_wmma_f32_16x16x32_bf16(false, aP, false, bv, (short)0, o[t], false, false);
        }
        __syncthreads();   // protect Kt/Vt before next stage
    }

    // normalize + store to attn buffer [b][n][1024] bf16
    float inv[8];
#pragma unroll
    for (int r = 0; r < 8; ++r) inv[r] = 1.0f / li[r];
#pragma unroll
    for (int t = 0; t < 4; ++t)
#pragma unroll
        for (int r = 0; r < 8; ++r) {
            int row = qRow + r + half * 8;
            int col = h * 64 + t * 16 + ln;
            Ao[((size_t)b * 2048 + row) * 1024 + col] = f32_to_bf16_bits(o[t][r] * inv[r]);
        }
}

extern "C" void kernel_launch(void* const* d_in, const int* in_sizes, int n_in,
                              void* d_out, int out_size, void* d_ws, size_t ws_size,
                              hipStream_t stream) {
    const float* x     = (const float*)d_in[0];  // [2,2048,1024]
    const float* w_qkv = (const float*)d_in[1];  // [1024,3072]
    const float* w_out = (const float*)d_in[2];  // [1024,1024]
    const float* b_out = (const float*)d_in[3];  // [1024]
    float* out = (float*)d_out;

    char* ws = (char*)d_ws;
    unsigned short* xb    = (unsigned short*)(ws);                 //  8 MB: 4096x1024
    unsigned short* wqkvb = (unsigned short*)(ws + (8u << 20));    //  6 MB: 1024x3072
    unsigned short* woutb = (unsigned short*)(ws + (14u << 20));   //  2 MB: 1024x1024
    unsigned short* qb    = (unsigned short*)(ws + (16u << 20));   //  8 MB: [32][2048][64]
    unsigned short* kb    = (unsigned short*)(ws + (24u << 20));   //  8 MB
    unsigned short* vb    = (unsigned short*)(ws + (32u << 20));   //  8 MB
    unsigned short* attnb = (unsigned short*)(ws + (40u << 20));   //  8 MB: 4096x1024

    const int NX = 4096 * 1024, NW = 1024 * 3072, NO = 1024 * 1024;
    cvt_bf16_kernel<<<(NX + 255) / 256, 256, 0, stream>>>(x, xb, NX);
    cvt_bf16_kernel<<<(NW + 255) / 256, 256, 0, stream>>>(w_qkv, wqkvb, NW);
    cvt_bf16_kernel<<<(NO + 255) / 256, 256, 0, stream>>>(w_out, woutb, NO);

    // qkv GEMM: [4096,1024] @ [1024,3072] -> scatter q/k/v (q scaled by 64^-0.5)
    gemm_bf16_kernel<0><<<dim3(4096 / 128, 3072 / 128), 256, 0, stream>>>(
        xb, wqkvb, 3072, 1024, qb, kb, vb, nullptr, nullptr, 0.125f);

    // flash attention: grid (2048/64 query blocks, 32 bh)
    attn_kernel<<<dim3(32, 32), 128, 0, stream>>>(qb, kb, vb, attnb);

    // output projection: [4096,1024] @ [1024,1024] + bias -> f32 out
    gemm_bf16_kernel<1><<<dim3(4096 / 128, 1024 / 128), 256, 0, stream>>>(
        attnb, woutb, 1024, 1024, nullptr, nullptr, nullptr, out, b_out, 1.0f);
}
